// PatientGATModel_75307956568204
// MI455X (gfx1250) — matrix operand order repaired and checked by reading, not compiled, not run on hardware
//
#include <hip/hip_runtime.h>
#include <stdint.h>

#define NNODES 20000
#define NEDGES 320000
#define NBATCH 64

typedef __attribute__((ext_vector_type(16))) __bf16 v16bf;
typedef __attribute__((ext_vector_type(8)))  float  v8f;

// ---------- helpers ----------
__device__ __forceinline__ unsigned short f2bf(float f) {
  uint32_t u = __float_as_uint(f);
  uint32_t r = (u + 0x7FFFu + ((u >> 16) & 1u)) >> 16;   // round-to-nearest-even
  return (unsigned short)r;
}
__device__ __forceinline__ uint32_t ordf(float f) {      // order-preserving f32->u32
  uint32_t b = __float_as_uint(f);
  return (b & 0x80000000u) ? ~b : (b | 0x80000000u);
}
__device__ __forceinline__ float unordf(uint32_t u) {
  uint32_t b = (u & 0x80000000u) ? (u ^ 0x80000000u) : ~u;
  return __uint_as_float(b);
}

// ---------- utility kernels ----------
__global__ void fill_u32(uint32_t* p, uint32_t v, int n) {
  int i = blockIdx.x * blockDim.x + threadIdx.x;
  if (i < n) p[i] = v;
}

__global__ void f32_to_bf16_kernel(const float* __restrict__ in,
                                   unsigned short* __restrict__ out, int n) {
  int i = blockIdx.x * blockDim.x + threadIdx.x;
  if (i < n) out[i] = f2bf(in[i]);
}

// W [K,Ncol] f32 (row-major)  ->  Wt [Ncol,K] bf16
__global__ void wt_bf16_kernel(const float* __restrict__ W,
                               unsigned short* __restrict__ Wt, int K, int Ncol) {
  int i = blockIdx.x * blockDim.x + threadIdx.x;
  if (i < K * Ncol) {
    int k = i / Ncol, n = i % Ncol;
    Wt[n * K + k] = f2bf(W[i]);
  }
}

// ---------- WMMA GEMM: C[M,Ncol] = A[M,K](bf16) x Bt[Ncol,K](bf16)^T, f32 accum ----------
// One wave computes a 16x64 strip (4 N-tiles) so the A fragment feeds 4 WMMAs.
// Fragment loads are contiguous 16B chunks -> global_load_b128.
__global__ void gemm_bf16_wmma(const unsigned short* __restrict__ A,
                               const unsigned short* __restrict__ Bt,
                               float* __restrict__ C, int M, int K, int Ncol) {
  int lane    = threadIdx.x;                              // 0..31 (wave32)
  int tile_n0 = blockIdx.x * 4;                           // first of 4 N-tiles
  int tile_m  = blockIdx.y * blockDim.y + threadIdx.y;    // wave-uniform
  if (tile_m * 16 >= M) return;                           // whole-wave guard (EXEC all-1)
  int half = lane >> 4;
  int lm   = lane & 15;
  long arow = (long)(tile_m * 16 + lm) * K;               // A row held by this lane

  union Frag { v16bf v; int4 q[2]; };
  v8f acc[4] = {};

#pragma unroll 2
  for (int kb = 0; kb < K; kb += 32) {
    Frag a;
    // A 16x32 bf16: VGPR 0..3 <- K = half*8 + 0..7 ; VGPR 4..7 <- K = 16 + half*8 + 0..7
    a.q[0] = *(const int4*)(A + arow + kb + half * 8);
    a.q[1] = *(const int4*)(A + arow + kb + 16 + half * 8);
    Frag b[4];
#pragma unroll
    for (int t = 0; t < 4; ++t) {
      // B 32x16 bf16: lane n = lane%16, VGPR 0..3 <- K = half*8 + 0..7 ; VGPR 4..7 <- K = 16 + half*8 + 0..7
      const unsigned short* bp = Bt + (long)((tile_n0 + t) * 16 + lm) * K + kb + half * 8;
      b[t].q[0] = *(const int4*)(bp);
      b[t].q[1] = *(const int4*)(bp + 16);
    }
#pragma unroll
    for (int t = 0; t < 4; ++t)
      acc[t] = __builtin_amdgcn_wmma_f32_16x16x32_bf16(false, a.v, false, b[t].v,
                                                       (short)0, acc[t], false, false);
  }

#pragma unroll
  for (int t = 0; t < 4; ++t) {
    int col = (tile_n0 + t) * 16 + lm;
#pragma unroll
    for (int r = 0; r < 8; ++r)   // C/D: VGPR r -> M = half*8 + r, N = lane%16
      C[(long)(tile_m * 16 + half * 8 + r) * Ncol + col] = acc[t][r];
  }
}

// ---------- per-node attention scores: s/d [N,heads] ----------
__global__ void scores_kernel(const float* __restrict__ h, const float* __restrict__ asrc,
                              const float* __restrict__ adst, float* __restrict__ s,
                              float* __restrict__ d, int n, int heads, int C) {
  int i = blockIdx.x * blockDim.x + threadIdx.x;
  if (i >= n * heads) return;
  int node = i / heads, hd = i % heads;
  const float* hp = h + (long)node * heads * C + hd * C;
  float ss = 0.f, dd = 0.f;
  for (int c = 0; c < C; ++c) { float v = hp[c]; ss += v * asrc[hd * C + c]; dd += v * adst[hd * C + c]; }
  s[i] = ss; d[i] = dd;
}

// ---------- edge pass 1: segment max (ordered-uint atomicMax) ----------
__global__ void edge_max_kernel(const int* __restrict__ ei, const float* __restrict__ s,
                                const float* __restrict__ d, uint32_t* __restrict__ m,
                                int E, int n, int heads) {
  int i = blockIdx.x * blockDim.x + threadIdx.x;
  int tot = (E + n) * heads;
  if (i >= tot) return;
  int e = i / heads, hd = i % heads;
  int src, dst;
  if (e < E) { src = ei[e]; dst = ei[E + e]; } else { src = dst = e - E; }
  float v = s[src * heads + hd] + d[dst * heads + hd];
  v = v > 0.f ? v : 0.2f * v;                             // leaky_relu(0.2)
  atomicMax(&m[dst * heads + hd], ordf(v));
}

// ---------- edge pass 2: exp-sum + alpha-weighted aggregation ----------
__global__ void edge_agg_kernel(const int* __restrict__ ei, const float* __restrict__ s,
                                const float* __restrict__ d, const uint32_t* __restrict__ m,
                                const float* __restrict__ h, float* __restrict__ outacc,
                                float* __restrict__ denom, int E, int n, int heads, int C) {
  int HC = heads * C;
  int i = blockIdx.x * blockDim.x + threadIdx.x;
  int tot = (E + n) * HC;
  if (i >= tot) return;
  int e = i / HC, c = i % HC, hd = c / C;
  int src, dst;
  if (e < E) { src = ei[e]; dst = ei[E + e]; } else { src = dst = e - E; }
  float v = s[src * heads + hd] + d[dst * heads + hd];
  v = v > 0.f ? v : 0.2f * v;
  float ex = expf(v - unordf(m[dst * heads + hd]));
  atomicAdd(&outacc[(long)dst * HC + c], ex * h[(long)src * HC + c]);
  if ((c % C) == 0) atomicAdd(&denom[dst * heads + hd], ex);
}

// ---------- finalize: /denom, +bias, ELU; write bf16 (next GEMM) and optional f32 ----------
__global__ void finalize_kernel(const float* __restrict__ outacc, const float* __restrict__ denom,
                                const float* __restrict__ bias, unsigned short* __restrict__ xbf,
                                float* __restrict__ xf32, int n, int heads, int C) {
  int HC = heads * C;
  int i = blockIdx.x * blockDim.x + threadIdx.x;
  if (i >= n * HC) return;
  int node = i / HC, c = i % HC, hd = c / C;
  float v = outacc[i] / (denom[node * heads + hd] + 1e-16f) + bias[c];
  v = v > 0.f ? v : (expf(v) - 1.0f);                     // ELU
  xbf[i] = f2bf(v);
  if (xf32) xf32[i] = v;
}

// ---------- batch mean pooling ----------
__global__ void pool_kernel(const float* __restrict__ feat, const int* __restrict__ batch,
                            float* __restrict__ pool, float* __restrict__ cnt, int n, int C) {
  int i = blockIdx.x * blockDim.x + threadIdx.x;
  if (i >= n * C) return;
  int node = i / C, c = i % C;
  int b = batch[node];
  atomicAdd(&pool[b * C + c], feat[i]);
  if (c == 0) atomicAdd(&cnt[b], 1.0f);
}
__global__ void pool_div_kernel(float* pool, const float* cnt, int B, int C) {
  int i = blockIdx.x * blockDim.x + threadIdx.x;
  if (i >= B * C) return;
  pool[i] /= fmaxf(cnt[i / C], 1.0f);
}

// ---------- tiny MLP heads (one block per batch element) ----------
__global__ void mlp_kernel(const float* __restrict__ poolL, const float* __restrict__ poolR,
                           const float* w1a, const float* b1a, const float* w2a, const float* b2a,
                           const float* w1b, const float* b1b, const float* w2b, const float* b2b,
                           float* __restrict__ out) {
  __shared__ float emb[128];
  __shared__ float h1[64];
  int b = blockIdx.x, t = threadIdx.x;                    // 64 threads
  emb[t] = poolL[b * 64 + t];
  emb[64 + t] = poolR[b * 64 + t];
  __syncthreads();
  float acc = b1a[t];
  for (int k = 0; k < 128; ++k) acc += emb[k] * w1a[k * 64 + t];
  h1[t] = fmaxf(acc, 0.f);
  __syncthreads();
  if (t < 2) {
    float l = b2a[t];
    for (int k = 0; k < 64; ++k) l += h1[k] * w2a[k * 2 + t];
    out[b * 2 + t] = l;                                   // logits_hc_pd
  }
  __syncthreads();
  acc = b1b[t];
  for (int k = 0; k < 128; ++k) acc += emb[k] * w1b[k * 64 + t];
  h1[t] = fmaxf(acc, 0.f);
  __syncthreads();
  if (t < 2) {
    float l = b2b[t];
    for (int k = 0; k < 64; ++k) l += h1[k] * w2b[k * 2 + t];
    out[128 + b * 2 + t] = l;                             // logits_pd_dd
  }
}

// ================= host side =================
static inline dim3 g1(long n, int b) { return dim3((unsigned)((n + b - 1) / b)); }

struct Scratch {
  unsigned short* xbf; float* h; float* outacc; float* s; float* d;
  uint32_t* m; float* denom;
  unsigned short *wt1, *wt2, *wt3;
  float *poolL, *poolR, *cntL, *cntR;
};

static void run_layer(hipStream_t stream, Scratch& ws, const unsigned short* wt,
                      const float* asrc, const float* adst, const float* bias,
                      const int* ei, int K, int Ncol, int heads, float* f32_out) {
  const int N = NNODES, E = NEDGES;
  const int C = Ncol / heads;
  gemm_bf16_wmma<<<dim3(Ncol / 64, (N / 16 + 3) / 4), dim3(32, 4), 0, stream>>>(
      ws.xbf, wt, ws.h, N, K, Ncol);
  scores_kernel<<<g1((long)N * heads, 256), 256, 0, stream>>>(ws.h, asrc, adst, ws.s, ws.d, N, heads, C);
  fill_u32<<<g1((long)N * heads, 256), 256, 0, stream>>>(ws.m, 0x007FFFFFu /* ord(-inf) */, N * heads);
  fill_u32<<<g1((long)N * heads, 256), 256, 0, stream>>>((uint32_t*)ws.denom, 0u, N * heads);
  fill_u32<<<g1((long)N * Ncol, 256), 256, 0, stream>>>((uint32_t*)ws.outacc, 0u, N * Ncol);
  edge_max_kernel<<<g1((long)(E + N) * heads, 256), 256, 0, stream>>>(ei, ws.s, ws.d, ws.m, E, N, heads);
  edge_agg_kernel<<<g1((long)(E + N) * Ncol, 256), 256, 0, stream>>>(
      ei, ws.s, ws.d, ws.m, ws.h, ws.outacc, ws.denom, E, N, heads, C);
  finalize_kernel<<<g1((long)N * Ncol, 256), 256, 0, stream>>>(
      ws.outacc, ws.denom, bias, ws.xbf, f32_out, N, heads, C);
}

static void run_graph(hipStream_t stream, Scratch& ws, const float* x, const int* ei,
                      const int* batch, float* pool, float* cnt,
                      const float* as1, const float* ad1, const float* b1,
                      const float* as2, const float* ad2, const float* b2,
                      const float* as3, const float* ad3, const float* b3) {
  const int N = NNODES;
  f32_to_bf16_kernel<<<g1((long)N * 128, 256), 256, 0, stream>>>(x, ws.xbf, N * 128);
  run_layer(stream, ws, ws.wt1, as1, ad1, b1, ei, 128, 256, 4, nullptr);
  run_layer(stream, ws, ws.wt2, as2, ad2, b2, ei, 256, 256, 4, nullptr);
  run_layer(stream, ws, ws.wt3, as3, ad3, b3, ei, 256, 64, 1, ws.h);  // f32 result -> ws.h[N*64]
  fill_u32<<<g1(NBATCH * 64, 256), 256, 0, stream>>>((uint32_t*)pool, 0u, NBATCH * 64);
  fill_u32<<<g1(NBATCH, 256), 256, 0, stream>>>((uint32_t*)cnt, 0u, NBATCH);
  pool_kernel<<<g1((long)N * 64, 256), 256, 0, stream>>>(ws.h, batch, pool, cnt, N, 64);
  pool_div_kernel<<<g1(NBATCH * 64, 256), 256, 0, stream>>>(pool, cnt, NBATCH, 64);
}

extern "C" void kernel_launch(void* const* d_in, const int* in_sizes, int n_in,
                              void* d_out, int out_size, void* d_ws, size_t ws_size,
                              hipStream_t stream) {
  const float* x_left   = (const float*)d_in[0];
  const float* x_right  = (const float*)d_in[1];
  const int*   ei_left  = (const int*)d_in[2];
  const int*   ei_right = (const int*)d_in[3];
  const int*   bat_left = (const int*)d_in[4];
  const int*   bat_right= (const int*)d_in[5];
  const float* w1 = (const float*)d_in[6];
  const float* as1= (const float*)d_in[7];
  const float* ad1= (const float*)d_in[8];
  const float* b1 = (const float*)d_in[9];
  const float* w2 = (const float*)d_in[10];
  const float* as2= (const float*)d_in[11];
  const float* ad2= (const float*)d_in[12];
  const float* b2 = (const float*)d_in[13];
  const float* w3 = (const float*)d_in[14];
  const float* as3= (const float*)d_in[15];
  const float* ad3= (const float*)d_in[16];
  const float* b3 = (const float*)d_in[17];
  const float* fc1_w1 = (const float*)d_in[18];
  const float* fc1_b1 = (const float*)d_in[19];
  const float* fc1_w2 = (const float*)d_in[20];
  const float* fc1_b2 = (const float*)d_in[21];
  const float* fc2_w1 = (const float*)d_in[22];
  const float* fc2_b1 = (const float*)d_in[23];
  const float* fc2_w2 = (const float*)d_in[24];
  const float* fc2_b2 = (const float*)d_in[25];
  float* out = (float*)d_out;

  // carve workspace
  char* base = (char*)d_ws;
  size_t off = 0;
  auto alloc = [&](size_t bytes) -> void* {
    void* p = base + off;
    off += (bytes + 255) & ~(size_t)255;
    return p;
  };
  const long N = NNODES;
  Scratch ws;
  ws.xbf    = (unsigned short*)alloc(N * 256 * 2);
  ws.h      = (float*)alloc(N * 256 * 4);
  ws.outacc = (float*)alloc(N * 256 * 4);
  ws.s      = (float*)alloc(N * 4 * 4);
  ws.d      = (float*)alloc(N * 4 * 4);
  ws.m      = (uint32_t*)alloc(N * 4 * 4);
  ws.denom  = (float*)alloc(N * 4 * 4);
  ws.wt1    = (unsigned short*)alloc(256 * 128 * 2);
  ws.wt2    = (unsigned short*)alloc(256 * 256 * 2);
  ws.wt3    = (unsigned short*)alloc(64 * 256 * 2);
  ws.poolL  = (float*)alloc(NBATCH * 64 * 4);
  ws.poolR  = (float*)alloc(NBATCH * 64 * 4);
  ws.cntL   = (float*)alloc(NBATCH * 4);
  ws.cntR   = (float*)alloc(NBATCH * 4);

  // transpose + quantize weights once (shared across both graphs)
  wt_bf16_kernel<<<g1(128 * 256, 256), 256, 0, stream>>>(w1, ws.wt1, 128, 256);
  wt_bf16_kernel<<<g1(256 * 256, 256), 256, 0, stream>>>(w2, ws.wt2, 256, 256);
  wt_bf16_kernel<<<g1(256 * 64, 256), 256, 0, stream>>>(w3, ws.wt3, 256, 64);

  run_graph(stream, ws, x_left, ei_left, bat_left, ws.poolL, ws.cntL,
            as1, ad1, b1, as2, ad2, b2, as3, ad3, b3);
  run_graph(stream, ws, x_right, ei_right, bat_right, ws.poolR, ws.cntR,
            as1, ad1, b1, as2, ad2, b2, as3, ad3, b3);

  mlp_kernel<<<NBATCH, 64, 0, stream>>>(ws.poolL, ws.poolR,
                                        fc1_w1, fc1_b1, fc1_w2, fc1_b2,
                                        fc2_w1, fc2_b1, fc2_w2, fc2_b2, out);
}
